// RestaurantGNN_11596411699472
// MI455X (gfx1250) — matrix operand order repaired and checked by reading, not compiled, not run on hardware
//
#include <hip/hip_runtime.h>
#include <stdint.h>

#define Nn   8192
#define Dd   128
#define Hh   256
#define DFFd 2048
#define HDd  64
#define QKVD 384            // qkv row stride (3*D)
#define Ee   (Nn * 32)
#define ENe  (Ee + Nn)
#define CHr  1024           // FFN row chunk

typedef __attribute__((ext_vector_type(16))) _Float16 v16h;
typedef __attribute__((ext_vector_type(8)))  float    v8f;

// ---------------- CDNA5 async global->LDS copy (ASYNCcnt) -------------------
__device__ __forceinline__ void async_ld128(unsigned ldsoff, const void* g) {
    asm volatile("global_load_async_to_lds_b128 %0, %1, off"
                 :: "v"(ldsoff), "v"((unsigned long long)(uintptr_t)g)
                 : "memory");
}
__device__ __forceinline__ void wait_async8() { asm volatile("s_wait_asynccnt 0x8" ::: "memory"); }
__device__ __forceinline__ void wait_async0() { asm volatile("s_wait_asynccnt 0x0" ::: "memory"); }

// ---------- monotone float<->uint encoding for atomicMax on floats ----------
__device__ __forceinline__ unsigned fenc(float f) {
    unsigned u = __float_as_uint(f);
    return (u & 0x80000000u) ? ~u : (u | 0x80000000u);
}
__device__ __forceinline__ float fdec(unsigned u) {
    unsigned b = (u & 0x80000000u) ? (u & 0x7fffffffu) : ~u;
    return __uint_as_float(b);
}

// ============================ AGNN conv =====================================
__global__ void k_invnorm(const float* __restrict__ x, float* __restrict__ invn) {
    int lane = threadIdx.x;
    int row  = blockIdx.x * blockDim.y + threadIdx.y;
    float4 v = *(const float4*)(x + (size_t)row * Dd + lane * 4);
    float ss = v.x * v.x + v.y * v.y + v.z * v.z + v.w * v.w;
    ss += __shfl_xor(ss, 1, 32);  ss += __shfl_xor(ss, 2, 32);
    ss += __shfl_xor(ss, 4, 32);  ss += __shfl_xor(ss, 8, 32);
    ss += __shfl_xor(ss, 16, 32);
    if (lane == 0) invn[row] = 1.0f / fmaxf(sqrtf(ss), 1e-12f);
}

__global__ void k_alpha(const float* __restrict__ x, const int* __restrict__ ei,
                        const float* __restrict__ invn, const float* __restrict__ betap,
                        float* __restrict__ alpha, unsigned* __restrict__ mmax) {
    int lane = threadIdx.x;
    long long e = (long long)blockIdx.x * blockDim.y + threadIdx.y;
    if (e >= ENe) return;
    int src, dst;
    if (e < Ee) { src = ei[e]; dst = ei[Ee + e]; }
    else        { src = dst = (int)(e - Ee); }
    float4 a = *(const float4*)(x + (size_t)src * Dd + lane * 4);
    float4 b = *(const float4*)(x + (size_t)dst * Dd + lane * 4);
    float d = a.x * b.x + a.y * b.y + a.z * b.z + a.w * b.w;
    d += __shfl_xor(d, 1, 32);  d += __shfl_xor(d, 2, 32);
    d += __shfl_xor(d, 4, 32);  d += __shfl_xor(d, 8, 32);
    d += __shfl_xor(d, 16, 32);
    if (lane == 0) {
        float al = betap[0] * invn[src] * invn[dst] * d;
        alpha[e] = al;
        atomicMax(&mmax[dst], fenc(al));
    }
}

__global__ void k_expz(const int* __restrict__ ei, float* __restrict__ alpha,
                       const unsigned* __restrict__ mmax, float* __restrict__ zsum) {
    long long e = (long long)blockIdx.x * blockDim.x + threadIdx.x;
    if (e >= ENe) return;
    int dst = (e < Ee) ? ei[Ee + e] : (int)(e - Ee);
    float ex = __expf(alpha[e] - fdec(mmax[dst]));
    alpha[e] = ex;
    atomicAdd(&zsum[dst], ex);
}

__global__ void k_agg(const float* __restrict__ x, const int* __restrict__ ei,
                      const float* __restrict__ alpha, const float* __restrict__ zsum,
                      float* __restrict__ hagg) {
    int lane = threadIdx.x;
    long long e = (long long)blockIdx.x * blockDim.y + threadIdx.y;
    if (e >= ENe) return;
    int src, dst;
    if (e < Ee) { src = ei[e]; dst = ei[Ee + e]; }
    else        { src = dst = (int)(e - Ee); }
    float w = alpha[e] / zsum[dst];
    float4 v = *(const float4*)(x + (size_t)src * Dd + lane * 4);
    float* o = hagg + (size_t)dst * Dd + lane * 4;
    atomicAdd(o + 0, w * v.x);
    atomicAdd(o + 1, w * v.y);
    atomicAdd(o + 2, w * v.z);
    atomicAdd(o + 3, w * v.w);
}

// ============================ f32 -> f16 convert ============================
__global__ void k_cvt(const float* __restrict__ src, _Float16* __restrict__ dst, int count) {
    int i = blockIdx.x * blockDim.x + threadIdx.x;
    if (i < count) dst[i] = (_Float16)src[i];
}

// ============================ WMMA GEMM (16x64/wave) ========================
// Y[M,Nout] = Xh[M,K](f16) @ Wh[Nout,K](f16)^T + bias. Optional relu.
// Outputs: Y (f32, nullable) and Yh (f16, nullable). One wave -> 16x64 tile.
__global__ void k_gemm4(const _Float16* __restrict__ Xh, const _Float16* __restrict__ Wh,
                        const float* __restrict__ bias, float* __restrict__ Y,
                        _Float16* __restrict__ Yh, int M, int K, int Nout, int relu) {
    int wid  = threadIdx.x >> 5;
    int lane = threadIdx.x & 31;
    int ntn  = Nout >> 6;                      // 64-wide column tiles
    int tile = blockIdx.x * 4 + wid;
    if (tile >= (M >> 4) * ntn) return;        // wave-uniform exit
    int nt = tile % ntn, mt = tile / ntn;
    int col  = lane & 15, half = lane >> 4;
    int arow  = (mt << 4) + col;
    int bcol0 = (nt << 6) + col;
    v8f acc0 = {}, acc1 = {}, acc2 = {}, acc3 = {};
    for (int k0 = 0; k0 < K; k0 += 32) {
        v16h a;
        const _Float16* ap = Xh + (size_t)arow * K + k0 + half * 16;
        if (k0 + 32 < K) __builtin_prefetch(ap + 32, 0, 3);
#pragma unroll
        for (int i = 0; i < 8; i++) { a[i] = ap[i]; a[i + 8] = ap[8 + i]; }
#define BSTEP(J, ACC)                                                              \
        {                                                                          \
            const _Float16* bp = Wh + (size_t)(bcol0 + 16 * J) * K + k0 + half*16; \
            v16h b;                                                                \
            _Pragma("unroll")                                                      \
            for (int i = 0; i < 16; i++) b[i] = bp[i];                             \
            ACC = __builtin_amdgcn_wmma_f32_16x16x32_f16(false, a, false, b,       \
                                                         (short)0, ACC, false, false); \
        }
        BSTEP(0, acc0)  BSTEP(1, acc1)  BSTEP(2, acc2)  BSTEP(3, acc3)
#undef BSTEP
    }
#pragma unroll
    for (int j = 0; j < 4; j++) {
        v8f acc = (j == 0) ? acc0 : (j == 1) ? acc1 : (j == 2) ? acc2 : acc3;
        int oc = bcol0 + 16 * j;
        float bv = bias ? bias[oc] : 0.0f;
#pragma unroll
        for (int r = 0; r < 8; r++) {
            int row = (mt << 4) + r + 8 * half;
            float v = acc[r] + bv;
            if (relu) v = fmaxf(v, 0.0f);
            if (Y)  Y [(size_t)row * Nout + oc] = v;
            if (Yh) Yh[(size_t)row * Nout + oc] = (_Float16)v;
        }
    }
}

// ===================== flash attention (wave = 16 Q rows) ===================
// Q/K/V read strided out of qkvh [N, 384] f16; V tiles staged via async->LDS.
__device__ __forceinline__ v8f s_wmma(v16h qa0, v16h qa1, const _Float16* kp) {
    v16h kf;
    v8f s = {};
#pragma unroll
    for (int i = 0; i < 16; i++) kf[i] = kp[i];
    s = __builtin_amdgcn_wmma_f32_16x16x32_f16(false, qa0, false, kf, (short)0, s, false, false);
#pragma unroll
    for (int i = 0; i < 16; i++) kf[i] = kp[32 + i];
    s = __builtin_amdgcn_wmma_f32_16x16x32_f16(false, qa1, false, kf, (short)0, s, false, false);
    return s;
}

__global__ void k_flash(const _Float16* __restrict__ qkvh, _Float16* __restrict__ Oh) {
    __shared__ __align__(16) _Float16 Plds[4][16 * 32];
    __shared__ __align__(16) _Float16 Vlds[4][2][32 * 64];   // double-buffered
    int w    = threadIdx.x >> 5;
    int lane = threadIdx.x & 31;
    int tile = blockIdx.x * 4 + w;        // 0..1023
    int head = tile >> 9;
    int mt   = tile & 511;
    const _Float16* Qb = qkvh + head * HDd;             // + n*384
    const _Float16* Kb = qkvh + Dd + head * HDd;
    const _Float16* Vb = qkvh + 2 * Dd + head * HDd;
    int col = lane & 15, half = lane >> 4;
    int qrow = mt * 16 + col;

    v16h qa0, qa1;                        // Q A-frags: d 0..31 / 32..63
    {
        const _Float16* qp = Qb + (size_t)qrow * QKVD + half * 16;
#pragma unroll
        for (int i = 0; i < 8; i++) {
            qa0[i] = qp[i];      qa0[i + 8] = qp[8 + i];
            qa1[i] = qp[32 + i]; qa1[i + 8] = qp[40 + i];
        }
    }
    float mst[8], lsum[8];
#pragma unroll
    for (int r = 0; r < 8; r++) { mst[r] = -1e30f; lsum[r] = 0.0f; }
    v8f acc0 = {}, acc1 = {}, acc2 = {}, acc3 = {};

    auto issueV = [&](int kb, int buf) {
        const _Float16* g = Vb + (size_t)(kb + lane) * QKVD;    // 64 f16 = 128B row
        unsigned lo = (unsigned)(uintptr_t)&Vlds[w][buf][lane * 64];
#pragma unroll
        for (int j = 0; j < 8; j++) async_ld128(lo + j * 16, g + j * 8);
    };

    issueV(0, 0);
    int buf = 0;
    for (int kb = 0; kb < Nn; kb += 32) {
        v8f s0 = s_wmma(qa0, qa1, Kb + (size_t)(kb + col) * QKVD + half * 16);
        v8f s1 = s_wmma(qa0, qa1, Kb + (size_t)(kb + 16 + col) * QKVD + half * 16);

        bool more = (kb + 32 < Nn);                      // wave-uniform
        if (more) issueV(kb + 32, buf ^ 1);              // prefetch next V tile

        const float qscale = 0.125f;                     // 1/sqrt(64)
#pragma unroll
        for (int r = 0; r < 8; r++) {
            float a0 = s0[r] * qscale, a1 = s1[r] * qscale;
            float rm = fmaxf(a0, a1);
            rm = fmaxf(rm, __shfl_xor(rm, 1, 32));
            rm = fmaxf(rm, __shfl_xor(rm, 2, 32));
            rm = fmaxf(rm, __shfl_xor(rm, 4, 32));
            rm = fmaxf(rm, __shfl_xor(rm, 8, 32));
            float mnew = fmaxf(mst[r], rm);
            float scl  = __expf(mst[r] - mnew);
            float e0 = __expf(a0 - mnew), e1 = __expf(a1 - mnew);
            float rs = e0 + e1;
            rs += __shfl_xor(rs, 1, 32);  rs += __shfl_xor(rs, 2, 32);
            rs += __shfl_xor(rs, 4, 32);  rs += __shfl_xor(rs, 8, 32);
            lsum[r] = lsum[r] * scl + rs;
            mst[r]  = mnew;
            acc0[r] *= scl; acc1[r] *= scl; acc2[r] *= scl; acc3[r] *= scl;
            int prow = r + 8 * half;      // C/D layout -> row-major P in LDS
            Plds[w][prow * 32 + col]      = (_Float16)e0;
            Plds[w][prow * 32 + 16 + col] = (_Float16)e1;
        }
        v16h pa;                          // P as A-fragment (16 rows x 32 keys)
        {
            const _Float16* pp = &Plds[w][col * 32 + half * 16];
#pragma unroll
            for (int i = 0; i < 8; i++) { pa[i] = pp[i]; pa[i + 8] = pp[8 + i]; }
        }
        if (more) wait_async8(); else wait_async0();     // current V tile resident
        const _Float16* vt = &Vlds[w][buf][0];
        v16h vb;
#pragma unroll
        for (int i = 0; i < 16; i++) vb[i] = vt[(half * 16 + i) * 64 + 0 * 16 + col];
        acc0 = __builtin_amdgcn_wmma_f32_16x16x32_f16(false, pa, false, vb, (short)0, acc0, false, false);
#pragma unroll
        for (int i = 0; i < 16; i++) vb[i] = vt[(half * 16 + i) * 64 + 1 * 16 + col];
        acc1 = __builtin_amdgcn_wmma_f32_16x16x32_f16(false, pa, false, vb, (short)0, acc1, false, false);
#pragma unroll
        for (int i = 0; i < 16; i++) vb[i] = vt[(half * 16 + i) * 64 + 2 * 16 + col];
        acc2 = __builtin_amdgcn_wmma_f32_16x16x32_f16(false, pa, false, vb, (short)0, acc2, false, false);
#pragma unroll
        for (int i = 0; i < 16; i++) vb[i] = vt[(half * 16 + i) * 64 + 3 * 16 + col];
        acc3 = __builtin_amdgcn_wmma_f32_16x16x32_f16(false, pa, false, vb, (short)0, acc3, false, false);
        buf ^= 1;
    }
#pragma unroll
    for (int r = 0; r < 8; r++) {
        int row = mt * 16 + r + 8 * half;
        float inv = 1.0f / lsum[r];
        _Float16* orow = Oh + (size_t)row * Dd + head * HDd;
        orow[0 * 16 + col] = (_Float16)(acc0[r] * inv);
        orow[1 * 16 + col] = (_Float16)(acc1[r] * inv);
        orow[2 * 16 + col] = (_Float16)(acc2[r] * inv);
        orow[3 * 16 + col] = (_Float16)(acc3[r] * inv);
    }
}

// ====================== residual + layernorm (wave/row) =====================
__global__ void k_resln(const float* __restrict__ A, const float* __restrict__ B,
                        const float* __restrict__ g, const float* __restrict__ b,
                        float* __restrict__ Y, _Float16* __restrict__ Yh) {
    int lane = threadIdx.x;
    int row  = blockIdx.x * blockDim.y + threadIdx.y;
    float4 a = *(const float4*)(A + (size_t)row * Dd + lane * 4);
    float4 c = *(const float4*)(B + (size_t)row * Dd + lane * 4);
    float v0 = a.x + c.x, v1 = a.y + c.y, v2 = a.z + c.z, v3 = a.w + c.w;
    float s = v0 + v1 + v2 + v3;
    s += __shfl_xor(s, 1, 32); s += __shfl_xor(s, 2, 32);
    s += __shfl_xor(s, 4, 32); s += __shfl_xor(s, 8, 32); s += __shfl_xor(s, 16, 32);
    float mu = s * (1.0f / Dd);
    float d0 = v0 - mu, d1 = v1 - mu, d2 = v2 - mu, d3 = v3 - mu;
    float q = d0 * d0 + d1 * d1 + d2 * d2 + d3 * d3;
    q += __shfl_xor(q, 1, 32); q += __shfl_xor(q, 2, 32);
    q += __shfl_xor(q, 4, 32); q += __shfl_xor(q, 8, 32); q += __shfl_xor(q, 16, 32);
    float inv = rsqrtf(q * (1.0f / Dd) + 1e-5f);
    int cix = lane * 4;
    float o0 = d0 * inv * g[cix + 0] + b[cix + 0];
    float o1 = d1 * inv * g[cix + 1] + b[cix + 1];
    float o2 = d2 * inv * g[cix + 2] + b[cix + 2];
    float o3 = d3 * inv * g[cix + 3] + b[cix + 3];
    if (Y) {
        float* o = Y + (size_t)row * Dd + cix;
        o[0] = o0; o[1] = o1; o[2] = o2; o[3] = o3;
    }
    if (Yh) {
        _Float16* oh = Yh + (size_t)row * Dd + cix;
        oh[0] = (_Float16)o0; oh[1] = (_Float16)o1;
        oh[2] = (_Float16)o2; oh[3] = (_Float16)o3;
    }
}

// ======================== tiny head: [N,256] -> [N,2] =======================
__global__ void k_head(const float* __restrict__ X, const float* __restrict__ W,
                       const float* __restrict__ b, float* __restrict__ out) {
    int n = blockIdx.x * blockDim.x + threadIdx.x;
    if (n >= Nn) return;
    float s0 = b[0], s1 = b[1];
    const float* xr = X + (size_t)n * Hh;
#pragma unroll 8
    for (int k = 0; k < Hh; k++) {
        float v = xr[k];
        s0 += v * W[k];
        s1 += v * W[Hh + k];
    }
    out[n * 2 + 0] = s0;
    out[n * 2 + 1] = s1;
}

// ============================= launcher =====================================
static inline size_t au(size_t x) { return (x + 255) & ~(size_t)255; }

extern "C" void kernel_launch(void* const* d_in, const int* in_sizes, int n_in,
                              void* d_out, int out_size, void* d_ws, size_t ws_size,
                              hipStream_t stream) {
    const float* x    = (const float*)d_in[0];
    const int*   ei   = (const int*)d_in[1];
    const float* beta = (const float*)d_in[2];
    const float* inpw = (const float*)d_in[3];
    const float* inpb = (const float*)d_in[4];
    const float* outw = (const float*)d_in[5];
    const float* outb = (const float*)d_in[6];
    const float* ln1g = (const float*)d_in[7];
    const float* ln1b = (const float*)d_in[8];
    const float* ff1w = (const float*)d_in[9];
    const float* ff1b = (const float*)d_in[10];
    const float* ff2w = (const float*)d_in[11];
    const float* ff2b = (const float*)d_in[12];
    const float* ln2g = (const float*)d_in[13];
    const float* ln2b = (const float*)d_in[14];
    const float* fcw  = (const float*)d_in[15];
    const float* fcb  = (const float*)d_in[16];
    const float* hw   = (const float*)d_in[17];
    const float* hb   = (const float*)d_in[18];
    float* out = (float*)d_out;

    char* p = (char*)d_ws;
    size_t off = 0;
    float*     invn   = (float*)(p + off);     off = au(off + (size_t)Nn * 4);
    unsigned*  mmax   = (unsigned*)(p + off);  off = au(off + (size_t)Nn * 4);
    float*     zsum   = (float*)(p + off);     off = au(off + (size_t)Nn * 4);
    float*     alpha  = (float*)(p + off);     off = au(off + (size_t)ENe * 4);
    float*     hagg   = (float*)(p + off);     off = au(off + (size_t)Nn * Dd * 4);
    _Float16*  haggh  = (_Float16*)(p + off);  off = au(off + (size_t)Nn * Dd * 2);
    _Float16*  qkvh   = (_Float16*)(p + off);  off = au(off + (size_t)Nn * QKVD * 2);
    _Float16*  oatth  = (_Float16*)(p + off);  off = au(off + (size_t)Nn * Dd * 2);
    float*     oproj  = (float*)(p + off);     off = au(off + (size_t)Nn * Dd * 4);
    float*     h1     = (float*)(p + off);     off = au(off + (size_t)Nn * Dd * 4);
    _Float16*  h1h    = (_Float16*)(p + off);  off = au(off + (size_t)Nn * Dd * 2);
    _Float16*  ffbufh = (_Float16*)(p + off);  off = au(off + (size_t)CHr * DFFd * 2);
    float*     ff2o   = (float*)(p + off);     off = au(off + (size_t)Nn * Dd * 4);
    _Float16*  h2h    = (_Float16*)(p + off);  off = au(off + (size_t)Nn * Dd * 2);
    float*     fco    = (float*)(p + off);     off = au(off + (size_t)Nn * Hh * 4);
    _Float16*  inpwh  = (_Float16*)(p + off);  off = au(off + (size_t)QKVD * Dd * 2);
    _Float16*  outwh  = (_Float16*)(p + off);  off = au(off + (size_t)Dd * Dd * 2);
    _Float16*  ff1wh  = (_Float16*)(p + off);  off = au(off + (size_t)DFFd * Dd * 2);
    _Float16*  ff2wh  = (_Float16*)(p + off);  off = au(off + (size_t)Dd * DFFd * 2);
    _Float16*  fcwh   = (_Float16*)(p + off);  off = au(off + (size_t)Hh * Dd * 2);
    (void)off; (void)ws_size; (void)n_in; (void)in_sizes; (void)out_size;

    hipMemsetAsync(mmax, 0, (size_t)Nn * 4, stream);          // fenc-min
    hipMemsetAsync(zsum, 0, (size_t)Nn * 4, stream);
    hipMemsetAsync(hagg, 0, (size_t)Nn * Dd * 4, stream);

    // ---- weights to f16 (small, once per call) ----
    k_cvt<<<(QKVD * Dd + 255) / 256, 256, 0, stream>>>(inpw, inpwh, QKVD * Dd);
    k_cvt<<<(Dd * Dd + 255) / 256, 256, 0, stream>>>(outw, outwh, Dd * Dd);
    k_cvt<<<(DFFd * Dd + 255) / 256, 256, 0, stream>>>(ff1w, ff1wh, DFFd * Dd);
    k_cvt<<<(Dd * DFFd + 255) / 256, 256, 0, stream>>>(ff2w, ff2wh, Dd * DFFd);
    k_cvt<<<(Hh * Dd + 255) / 256, 256, 0, stream>>>(fcw, fcwh, Hh * Dd);

    // ---- AGNN conv ----
    k_invnorm<<<Nn / 8, dim3(32, 8), 0, stream>>>(x, invn);
    k_alpha<<<ENe / 8, dim3(32, 8), 0, stream>>>(x, ei, invn, beta, alpha, mmax);
    k_expz<<<(ENe + 255) / 256, 256, 0, stream>>>(ei, alpha, mmax, zsum);
    k_agg<<<ENe / 8, dim3(32, 8), 0, stream>>>(x, ei, alpha, zsum, hagg);
    k_cvt<<<(Nn * Dd + 255) / 256, 256, 0, stream>>>(hagg, haggh, Nn * Dd);

    // ---- in_proj: qkvh = hagg @ inpw^T + inpb (f16 out) ----
    {
        int tiles = (Nn / 16) * (QKVD / 64);
        k_gemm4<<<(tiles + 3) / 4, 128, 0, stream>>>(haggh, inpwh, inpb, nullptr, qkvh,
                                                     Nn, Dd, QKVD, 0);
    }

    // ---- flash attention (2 heads x 512 q-tiles, async V staging) ----
    k_flash<<<256, 128, 0, stream>>>(qkvh, oatth);

    // ---- out_proj + residual LN1 ----
    {
        int tiles = (Nn / 16) * (Dd / 64);
        k_gemm4<<<(tiles + 3) / 4, 128, 0, stream>>>(oatth, outwh, outb, oproj, nullptr,
                                                     Nn, Dd, Dd, 0);
    }
    k_resln<<<Nn / 8, dim3(32, 8), 0, stream>>>(hagg, oproj, ln1g, ln1b, h1, h1h);

    // ---- FFN, chunked over rows to bound scratch ----
    for (int c = 0; c < Nn / CHr; c++) {
        const _Float16* xin = h1h + (size_t)c * CHr * Dd;
        float* yout = ff2o + (size_t)c * CHr * Dd;
        int t1 = (CHr / 16) * (DFFd / 64);
        k_gemm4<<<(t1 + 3) / 4, 128, 0, stream>>>(xin, ff1wh, ff1b, nullptr, ffbufh,
                                                  CHr, Dd, DFFd, 1);
        int t2 = (CHr / 16) * (Dd / 64);
        k_gemm4<<<(t2 + 3) / 4, 128, 0, stream>>>(ffbufh, ff2wh, ff2b, yout, nullptr,
                                                  CHr, DFFd, Dd, 0);
    }
    k_resln<<<Nn / 8, dim3(32, 8), 0, stream>>>(h1, ff2o, ln2g, ln2b, nullptr, h2h);

    // ---- fc (relu) + head ----
    {
        int tiles = (Nn / 16) * (Hh / 64);
        k_gemm4<<<(tiles + 3) / 4, 128, 0, stream>>>(h2h, fcwh, fcb, fco, nullptr,
                                                     Nn, Dd, Hh, 1);
    }
    k_head<<<(Nn + 255) / 256, 256, 0, stream>>>(fco, hw, hb, out);
}
